// MultiHeadAttention_18408229831056
// MI455X (gfx1250) — matrix operand order repaired and checked by reading, not compiled, not run on hardware
//
#include <hip/hip_runtime.h>
#include <hip/hip_bf16.h>

typedef __bf16 bf16;
typedef __attribute__((ext_vector_type(16))) __bf16 v16bf;
typedef __attribute__((ext_vector_type(8)))  __bf16 v8bf;
typedef __attribute__((ext_vector_type(8)))  float  v8f;

#define D_MODEL 1024
#define N_HEADS 16
#define HEAD_DIM 64
#define SEQ 2048
#define BATCH 2

// ---------------------------------------------------------------------------
// Fragment loaders matching the CDNA5 WMMA VGPR layouts (CDNA5 ISA)
// ---------------------------------------------------------------------------

// A-fragment 16x32 bf16 from row-major [16 x >=32] source.
__device__ __forceinline__ v16bf load_a_frag(const bf16* base, int stride) {
  int lane = threadIdx.x & 31;
  int row  = lane & 15;
  int kb   = (lane & 16) >> 1;            // 0 or 8
  const bf16* p = base + row * stride + kb;
  v8bf lo = *(const v8bf*)(p);            // K = kb .. kb+7   (b128)
  v8bf hi = *(const v8bf*)(p + 16);       // K = kb+16..kb+23 (b128)
  v16bf r;
#pragma unroll
  for (int i = 0; i < 8; ++i) { r[i] = lo[i]; r[i + 8] = hi[i]; }
  return r;
}

// B-fragment 32x16 bf16; for each output column n, K contiguous at base+n*stride.
__device__ __forceinline__ v16bf load_b_frag(const bf16* base, int stride) {
  int lane = threadIdx.x & 31;
  int col  = lane & 15;
  int koff = (lane & 16) >> 1;            // 0 or 8
  const bf16* p = base + col * stride + koff;
  v8bf lo = *(const v8bf*)(p);
  v8bf hi = *(const v8bf*)(p + 16);
  v16bf r;
#pragma unroll
  for (int i = 0; i < 8; ++i) { r[i] = lo[i]; r[i + 8] = hi[i]; }
  return r;
}

// ---------------------------------------------------------------------------
// Ping-pong pipelined 32(M) x 64(N) x 1024(K) tile.
// Two fragment buffer sets alternate: no cross-buffer register copies, so no
// WMMA->VALU WAR hazard NOPs, and loads stay in flight under the WMMAs.
// ---------------------------------------------------------------------------
__device__ __forceinline__ void load_ab(const bf16* __restrict__ A,
                                        const bf16* __restrict__ W,
                                        int n0, int kk,
                                        v16bf a[2], v16bf b[4]) {
#pragma unroll
  for (int mi = 0; mi < 2; ++mi)
    a[mi] = load_a_frag(A + mi * 16 * D_MODEL + kk, D_MODEL);
#pragma unroll
  for (int j = 0; j < 4; ++j)
    b[j] = load_b_frag(W + (n0 + 16 * j) * D_MODEL + kk, D_MODEL);
}

__device__ __forceinline__ void wmma8(const v16bf a[2], const v16bf b[4], v8f acc[2][4]) {
#pragma unroll
  for (int mi = 0; mi < 2; ++mi)
#pragma unroll
    for (int j = 0; j < 4; ++j)
      acc[mi][j] = __builtin_amdgcn_wmma_f32_16x16x32_bf16(
          false, a[mi], false, b[j], (short)0, acc[mi][j], false, false);
}

__device__ __forceinline__ void gemm_tile_32x64(const bf16* __restrict__ A,   // at row m0, ld=1024
                                                const bf16* __restrict__ W,   // weight base, ld=1024
                                                int n0, v8f acc[2][4]) {
  v16bf a0[2], b0[4], a1[2], b1[4];
  load_ab(A, W, n0, 0, a0, b0);
  for (int kk = 0; kk < D_MODEL - 64; kk += 64) {
    load_ab(A, W, n0, kk + 32, a1, b1);   // prefetch odd half
    wmma8(a0, b0, acc);                   // consume even half
    load_ab(A, W, n0, kk + 64, a0, b0);   // prefetch next even half
    wmma8(a1, b1, acc);                   // consume odd half
  }
  load_ab(A, W, n0, D_MODEL - 32, a1, b1);
  wmma8(a0, b0, acc);                     // k = D_MODEL-64
  wmma8(a1, b1, acc);                     // k = D_MODEL-32
}

// ---------------------------------------------------------------------------
// fp32 -> bf16 conversion
// ---------------------------------------------------------------------------
__global__ void cvt_f32_bf16(const float* __restrict__ in, bf16* __restrict__ out, int n) {
  int i = blockIdx.x * blockDim.x + threadIdx.x;
  if (i < n) out[i] = (bf16)in[i];
}

// ---------------------------------------------------------------------------
// QKV projection GEMM with fused RoPE epilogue. blockIdx.y: 0=Q, 1=K, 2=V.
// One wave -> 32x64 tile. 2048 waves per projection.
// ---------------------------------------------------------------------------
__global__ void __launch_bounds__(256) qkv_rope_gemm(
    const bf16* __restrict__ xb,
    const bf16* __restrict__ Wqb, const bf16* __restrict__ Wkb,
    const bf16* __restrict__ Wvb,
    bf16* __restrict__ Qb, bf16* __restrict__ Kb, bf16* __restrict__ VbT,
    float* __restrict__ kcache, float* __restrict__ vcache) {
  const int proj = blockIdx.y;
  const bf16* W = (proj == 0) ? Wqb : (proj == 1) ? Wkb : Wvb;

  int wave = (blockIdx.x * blockDim.x + threadIdx.x) >> 5;   // 0..2047
  int mt = wave >> 4;                     // 128 M tiles of 32
  int nt = wave & 15;                     // 16 N groups of 64
  int m0 = mt * 32;
  int n0 = nt * 64;
  int lane = threadIdx.x & 31;

  v8f acc[2][4] = {};
  gemm_tile_32x64(xb + m0 * D_MODEL, W, n0, acc);

  int half8 = (lane & 16) >> 1;
#pragma unroll
  for (int mi = 0; mi < 2; ++mi) {
#pragma unroll
    for (int j = 0; j < 4; ++j) {
      int col = n0 + 16 * j + (lane & 15);
      int h = col >> 6;                   // head
      int c = col & 63;                   // dim within head
#pragma unroll
      for (int i = 0; i < 8; ++i) {
        int row = m0 + mi * 16 + half8 + i;
        int s = row & (SEQ - 1);
        int bb = row >> 11;               // batch
        int bh = bb * N_HEADS + h;
        float v = acc[mi][j][i];
        if (proj < 2) {
          // RoPE: even/odd interleaved pair lives in adjacent lanes
          float partner = __shfl_xor(v, 1, 32);
          int p = c >> 1;
          float inv = __expf(-(float)p * 0.28782313662425572f);  // ln(10000)/32
          float sn, cs;
          __sincosf((float)s * inv, &sn, &cs);
          float r = (c & 1) ? (partner * sn + v * cs)
                            : (v * cs - partner * sn);
          int idx = (bh * SEQ + s) * HEAD_DIM + c;               // [B,H,S,hd]
          if (proj == 0) {
            Qb[idx] = (bf16)r;
          } else {
            Kb[idx] = (bf16)r;
            kcache[idx] = r;
          }
        } else {
          int idx = (bh * SEQ + s) * HEAD_DIM + c;
          vcache[idx] = v;
          VbT[(bh * HEAD_DIM + c) * SEQ + s] = (bf16)v;          // [B,H,hd,S]
        }
      }
    }
  }
}

// ---------------------------------------------------------------------------
// Flash attention: one wave per (b, h, 16-query tile); 32-key blocks, causal.
// K frags loaded first, V frags issued immediately after so they complete
// under the softmax VALU work.
// ---------------------------------------------------------------------------
__global__ void __launch_bounds__(256) attn_kernel(
    const bf16* __restrict__ Qb, const bf16* __restrict__ Kb,
    const bf16* __restrict__ VbT, bf16* __restrict__ Ob) {
  __shared__ bf16 psh[8 * 16 * 40];       // per-wave 16x32 P tile, stride 40
  int waveInBlk = threadIdx.x >> 5;
  int lane = threadIdx.x & 31;
  int wave = blockIdx.x * 8 + waveInBlk;  // 0..4095
  int bh = wave >> 7;                     // (b*H + h)
  int qt = wave & 127;
  int m0 = qt * 16;
  const float scale = 0.125f;             // 1/sqrt(64)
  int half8 = (lane & 16) >> 1;
  bf16* pw = psh + waveInBlk * (16 * 40);

  const bf16* qbase = Qb + (bh * SEQ + m0) * HEAD_DIM;
  v16bf aq0 = load_a_frag(qbase, HEAD_DIM);        // K-dim 0..31 of hd
  v16bf aq1 = load_a_frag(qbase + 32, HEAD_DIM);   // K-dim 32..63

  v8f o[4] = {};
  float mrow[8], lrow[8];
#pragma unroll
  for (int i = 0; i < 8; ++i) { mrow[i] = -1e30f; lrow[i] = 0.f; }

  for (int j0 = 0; j0 < m0 + 16; j0 += 32) {
    // ---- issue K-fragment loads, then V-fragment loads (in-order completion
    //      means QK WMMAs only drain the K loads) ----
    v16bf bk[2][2];
#pragma unroll
    for (int t = 0; t < 2; ++t) {
      const bf16* kb0 = Kb + (bh * SEQ + j0 + 16 * t) * HEAD_DIM;
      bk[t][0] = load_b_frag(kb0, HEAD_DIM);       // K[key, 0..31]
      bk[t][1] = load_b_frag(kb0 + 32, HEAD_DIM);  // K[key, 32..63]
    }
    v16bf bv[4];
#pragma unroll
    for (int dt = 0; dt < 4; ++dt) {
      const bf16* vb = VbT + (bh * HEAD_DIM + 16 * dt) * SEQ + j0;
      bv[dt] = load_b_frag(vb, SEQ);               // V^T: keys contiguous per dim
    }
    // ---- S = Q K^T ----
    v8f sA[2];
#pragma unroll
    for (int t = 0; t < 2; ++t) {
      v8f z = {};
      z = __builtin_amdgcn_wmma_f32_16x16x32_bf16(false, aq0, false, bk[t][0], (short)0, z, false, false);
      z = __builtin_amdgcn_wmma_f32_16x16x32_bf16(false, aq1, false, bk[t][1], (short)0, z, false, false);
      sA[t] = z;
    }
    // ---- scale + causal mask + row max ----
    float rmax[8];
#pragma unroll
    for (int i = 0; i < 8; ++i) rmax[i] = -1e30f;
#pragma unroll
    for (int t = 0; t < 2; ++t) {
      int key = j0 + 16 * t + (lane & 15);
#pragma unroll
      for (int i = 0; i < 8; ++i) {
        int q = m0 + half8 + i;
        float sc = sA[t][i] * scale;
        sc = (key > q) ? -1e30f : sc;
        sA[t][i] = sc;
        rmax[i] = fmaxf(rmax[i], sc);
      }
    }
#pragma unroll
    for (int i = 0; i < 8; ++i) {
      rmax[i] = fmaxf(rmax[i], __shfl_xor(rmax[i], 1, 32));
      rmax[i] = fmaxf(rmax[i], __shfl_xor(rmax[i], 2, 32));
      rmax[i] = fmaxf(rmax[i], __shfl_xor(rmax[i], 4, 32));
      rmax[i] = fmaxf(rmax[i], __shfl_xor(rmax[i], 8, 32));
    }
    // ---- online softmax update ----
    float corr[8], rsum[8];
#pragma unroll
    for (int i = 0; i < 8; ++i) {
      float mnew = fmaxf(mrow[i], rmax[i]);
      corr[i] = __expf(mrow[i] - mnew);
      mrow[i] = mnew;
      rsum[i] = 0.f;
    }
#pragma unroll
    for (int t = 0; t < 2; ++t) {
#pragma unroll
      for (int i = 0; i < 8; ++i) {
        float p = __expf(sA[t][i] - mrow[i]);
        sA[t][i] = p;
        rsum[i] += p;
      }
    }
#pragma unroll
    for (int i = 0; i < 8; ++i) {
      rsum[i] += __shfl_xor(rsum[i], 1, 32);
      rsum[i] += __shfl_xor(rsum[i], 2, 32);
      rsum[i] += __shfl_xor(rsum[i], 4, 32);
      rsum[i] += __shfl_xor(rsum[i], 8, 32);
      lrow[i] = lrow[i] * corr[i] + rsum[i];
    }
#pragma unroll
    for (int dt = 0; dt < 4; ++dt)
#pragma unroll
      for (int i = 0; i < 8; ++i)
        o[dt][i] *= corr[i];
    // ---- transpose P (C layout -> A layout) through LDS ----
#pragma unroll
    for (int t = 0; t < 2; ++t) {
      int col = 16 * t + (lane & 15);
#pragma unroll
      for (int i = 0; i < 8; ++i)
        pw[(half8 + i) * 40 + col] = (bf16)sA[t][i];
    }
    asm volatile("s_wait_dscnt 0" ::: "memory");   // single-wave LDS RAW fence
    v16bf ap = load_a_frag(pw, 40);
    // ---- O += P @ V_block ----
#pragma unroll
    for (int dt = 0; dt < 4; ++dt)
      o[dt] = __builtin_amdgcn_wmma_f32_16x16x32_bf16(
          false, ap, false, bv[dt], (short)0, o[dt], false, false);
  }

  // ---- normalize and write O in [B,S,D] (bf16) for the output GEMM ----
  float invl[8];
#pragma unroll
  for (int i = 0; i < 8; ++i) invl[i] = 1.0f / lrow[i];
  int bb = bh >> 4;
  int h  = bh & 15;
#pragma unroll
  for (int dt = 0; dt < 4; ++dt) {
    int col = h * HEAD_DIM + 16 * dt + (lane & 15);
#pragma unroll
    for (int i = 0; i < 8; ++i) {
      int srow = m0 + half8 + i;
      Ob[(bb * SEQ + srow) * D_MODEL + col] = (bf16)(o[dt][i] * invl[i]);
    }
  }
}

// ---------------------------------------------------------------------------
// Output projection GEMM: out = O @ Wo^T (fp32 straight to d_out)
// ---------------------------------------------------------------------------
__global__ void __launch_bounds__(256) out_gemm(
    const bf16* __restrict__ Ob, const bf16* __restrict__ Wob,
    float* __restrict__ out) {
  int wave = (blockIdx.x * blockDim.x + threadIdx.x) >> 5;   // 0..2047
  int mt = wave >> 4, nt = wave & 15;
  int m0 = mt * 32, n0 = nt * 64;
  int lane = threadIdx.x & 31;

  v8f acc[2][4] = {};
  gemm_tile_32x64(Ob + m0 * D_MODEL, Wob, n0, acc);

  int half8 = (lane & 16) >> 1;
#pragma unroll
  for (int mi = 0; mi < 2; ++mi)
#pragma unroll
    for (int j = 0; j < 4; ++j) {
      int col = n0 + 16 * j + (lane & 15);
#pragma unroll
      for (int i = 0; i < 8; ++i) {
        int row = m0 + mi * 16 + half8 + i;
        out[row * D_MODEL + col] = acc[mi][j][i];
      }
    }
}

// ---------------------------------------------------------------------------
extern "C" void kernel_launch(void* const* d_in, const int* in_sizes, int n_in,
                              void* d_out, int out_size, void* d_ws, size_t ws_size,
                              hipStream_t stream) {
  const float* x  = (const float*)d_in[0];
  // d_in[1] = mask (causal triu, reproduced analytically in-kernel)
  const float* Wq = (const float*)d_in[2];
  const float* Wk = (const float*)d_in[3];
  const float* Wv = (const float*)d_in[4];
  const float* Wo = (const float*)d_in[5];

  float* out    = (float*)d_out;                          // [B,S,D]
  float* kcache = out + (size_t)BATCH * SEQ * D_MODEL;    // [B,H,S,hd]
  float* vcache = kcache + (size_t)BATCH * N_HEADS * SEQ * HEAD_DIM;

  const size_t XN = (size_t)BATCH * SEQ * D_MODEL;        // 4,194,304
  const size_t WN = (size_t)D_MODEL * D_MODEL;            // 1,048,576
  bf16* xb  = (bf16*)d_ws;
  bf16* Wqb = xb + XN;
  bf16* Wkb = Wqb + WN;
  bf16* Wvb = Wkb + WN;
  bf16* Wob = Wvb + WN;
  bf16* Qb  = Wob + WN;   // [B,H,S,hd] bf16 (roped)
  bf16* Kb  = Qb + XN;    // [B,H,S,hd] bf16 (roped)
  bf16* VbT = Kb + XN;    // [B,H,hd,S] bf16
  bf16* Ob  = VbT + XN;   // [B,S,D]    bf16

  cvt_f32_bf16<<<XN / 256, 256, 0, stream>>>(x,  xb,  (int)XN);
  cvt_f32_bf16<<<WN / 256, 256, 0, stream>>>(Wq, Wqb, (int)WN);
  cvt_f32_bf16<<<WN / 256, 256, 0, stream>>>(Wk, Wkb, (int)WN);
  cvt_f32_bf16<<<WN / 256, 256, 0, stream>>>(Wv, Wvb, (int)WN);
  cvt_f32_bf16<<<WN / 256, 256, 0, stream>>>(Wo, Wob, (int)WN);

  qkv_rope_gemm<<<dim3(256, 3), 256, 0, stream>>>(xb, Wqb, Wkb, Wvb,
                                                  Qb, Kb, VbT, kcache, vcache);
  attn_kernel<<<512, 256, 0, stream>>>(Qb, Kb, VbT, Ob);
  out_gemm<<<256, 256, 0, stream>>>(Ob, Wob, out);
}
